// PromptPool_46093589021391
// MI455X (gfx1250) — matrix operand order repaired and checked by reading, not compiled, not run on hardware
//
#include <hip/hip_runtime.h>
#include <hip/hip_bf16.h>
#include <stdint.h>

#define B_N 8192
#define P_N 4096
#define D_N 1024
#define L_N 5
#define K_N 5
#define EPS_F 1e-8f

typedef __attribute__((ext_vector_type(16))) __bf16 v16bf;
typedef __attribute__((ext_vector_type(8)))  float  v8f;

union BFrag {
    v16bf v;
    uint4 q[2];
    unsigned short s[16];
};

__device__ __forceinline__ unsigned short f2bf_trunc(float f) {
    return (unsigned short)(__builtin_bit_cast(unsigned int, f) >> 16);
}
__device__ __forceinline__ float bf2f(unsigned short h) {
    return __builtin_bit_cast(float, ((unsigned int)h) << 16);
}
__device__ __forceinline__ bool better(float v1, int i1, float v2, int i2) {
    return (v1 < v2) || (v1 == v2 && i1 < i2);
}

// ---------------------------------------------------------------------------
// Split f32 -> (hi bf16, lo bf16), truncating split: hi + lo ~= f (>= 16 significant bits)
// One float4 per thread.
// ---------------------------------------------------------------------------
__global__ void PromptPool_split_kernel(const float* __restrict__ src,
                                        unsigned short* __restrict__ hi,
                                        unsigned short* __restrict__ lo,
                                        long n4) {
    long id = (long)blockIdx.x * blockDim.x + threadIdx.x;
    if (id >= n4) return;
    const float4 f = ((const float4*)src)[id];
    union { uint2 u; unsigned short s[4]; } h, l;
    float x;
    x = f.x; h.s[0] = f2bf_trunc(x); l.s[0] = f2bf_trunc(x - bf2f(h.s[0]));
    x = f.y; h.s[1] = f2bf_trunc(x); l.s[1] = f2bf_trunc(x - bf2f(h.s[1]));
    x = f.z; h.s[2] = f2bf_trunc(x); l.s[2] = f2bf_trunc(x - bf2f(h.s[2]));
    x = f.w; h.s[3] = f2bf_trunc(x); l.s[3] = f2bf_trunc(x - bf2f(h.s[3]));
    ((uint2*)hi)[id] = h.u;
    ((uint2*)lo)[id] = l.u;
}

// ---------------------------------------------------------------------------
// Row L2 norms: one wave32 per row. Rows [0,B) -> xn from x, [B,B+P) -> kn.
// ---------------------------------------------------------------------------
__global__ void PromptPool_norms_kernel(const float* __restrict__ x,
                                        const float* __restrict__ keys,
                                        float* __restrict__ xn,
                                        float* __restrict__ kn) {
    const int wv = threadIdx.x >> 5;
    const int lane = threadIdx.x & 31;
    const int row = blockIdx.x * 8 + wv;
    const float* src;
    float* dst;
    if (row < B_N)             { src = x    + (size_t)row * D_N;          dst = xn + row; }
    else if (row < B_N + P_N)  { src = keys + (size_t)(row - B_N) * D_N;  dst = kn + (row - B_N); }
    else return;
    const float4* s4 = (const float4*)src;
    float sum = 0.f;
    #pragma unroll
    for (int i = lane; i < D_N / 4; i += 32) {
        float4 f = s4[i];
        sum += f.x * f.x + f.y * f.y + f.z * f.z + f.w * f.w;
    }
    #pragma unroll
    for (int m = 16; m > 0; m >>= 1) sum += __shfl_xor(sum, m, 32);
    if (lane == 0) *dst = sqrtf(sum);
}

// ---------------------------------------------------------------------------
// Split-bf16 WMMA GEMM + cosine-distance epilogue.
// Block = 128 threads (4 waves). Block tile: 16 rows x 256 cols.
// Wave tile: 16 x 64 (4 independent 16x16 accumulator chains).
// A (x) staged hi/lo in LDS with padded stride (conflict-free ds_load_b128).
// B (keys) per-lane global_load_b128 from pre-split bf16 (L2-resident).
// dot = Ah*Bh + Ah*Bl + Al*Bh + Al*Bl  (f32 accumulate; ~f32-exact)
// ---------------------------------------------------------------------------
#define ASTRIDE 1032  // 1024 + 8 bf16 pad -> rows land on distinct LDS banks

__global__ __launch_bounds__(128) void PromptPool_gemm_dist_kernel(
        const unsigned short* __restrict__ xhi,
        const unsigned short* __restrict__ xlo,
        const unsigned short* __restrict__ khi,
        const unsigned short* __restrict__ klo,
        const float* __restrict__ xn,
        const float* __restrict__ kn,
        float* __restrict__ dist) {
    __shared__ alignas(16) unsigned short sAhi[16 * ASTRIDE];
    __shared__ alignas(16) unsigned short sAlo[16 * ASTRIDE];

    const int t = threadIdx.x;
    const int wv = t >> 5;
    const int lane = t & 31;
    const int m = lane & 15;         // row-in-tile (A) / col-in-tile (B,C)
    const int half = lane >> 4;
    const int koff = half * 16;      // A-fragment K sub-offset per lane half

    const int rowbase = blockIdx.x * 16;
    const int colbase0 = blockIdx.y * 256 + wv * 64;

    // Stage 16x1024 hi/lo A tile into LDS (source rows are contiguous).
    {
        const uint4* xh4 = (const uint4*)xhi;  // 8 bf16 per uint4
        const uint4* xl4 = (const uint4*)xlo;
        const size_t sbase = (size_t)rowbase * 128;  // uint4 units
        for (int i = t; i < 2048; i += 128) {
            const int r = i >> 7, c = i & 127;
            *((uint4*)(sAhi + r * ASTRIDE) + c) = xh4[sbase + i];
            *((uint4*)(sAlo + r * ASTRIDE) + c) = xl4[sbase + i];
        }
    }
    __syncthreads();

    // B-fragment base pointers (lane = column, K contiguous per lane half).
    const unsigned short* bhp[4];
    const unsigned short* blp[4];
    #pragma unroll
    for (int ct = 0; ct < 4; ++ct) {
        const size_t p = (size_t)(colbase0 + ct * 16 + m);
        bhp[ct] = khi + p * D_N + half * 16;
        blp[ct] = klo + p * D_N + half * 16;
    }

    v8f acc[4] = {};
    const unsigned short* aH = sAhi + m * ASTRIDE + koff;
    const unsigned short* aL = sAlo + m * ASTRIDE + koff;

    for (int k0 = 0; k0 < D_N; k0 += 32) {
        BFrag ah, al;
        ah.q[0] = *(const uint4*)(aH + k0);
        ah.q[1] = *(const uint4*)(aH + k0 + 8);
        al.q[0] = *(const uint4*)(aL + k0);
        al.q[1] = *(const uint4*)(aL + k0 + 8);
        #pragma unroll
        for (int ct = 0; ct < 4; ++ct) {
            BFrag bh, bl;
            bh.q[0] = *(const uint4*)(bhp[ct] + k0);
            bh.q[1] = *(const uint4*)(bhp[ct] + k0 + 8);
            bl.q[0] = *(const uint4*)(blp[ct] + k0);
            bl.q[1] = *(const uint4*)(blp[ct] + k0 + 8);
            acc[ct] = __builtin_amdgcn_wmma_f32_16x16x32_bf16(
                false, ah.v, false, bh.v, (short)0, acc[ct], false, false);
            acc[ct] = __builtin_amdgcn_wmma_f32_16x16x32_bf16(
                false, ah.v, false, bl.v, (short)0, acc[ct], false, false);
            acc[ct] = __builtin_amdgcn_wmma_f32_16x16x32_bf16(
                false, al.v, false, bh.v, (short)0, acc[ct], false, false);
            acc[ct] = __builtin_amdgcn_wmma_f32_16x16x32_bf16(
                false, al.v, false, bl.v, (short)0, acc[ct], false, false);
        }
    }

    // Epilogue: dist = 1 - dot / max(|x||k|, eps).  C layout: lanes 0-15 rows
    // 0..7, lanes 16-31 rows 8..15; N = lane%16.
    float xnv[8];
    #pragma unroll
    for (int r = 0; r < 8; ++r) xnv[r] = xn[rowbase + r + 8 * half];
    #pragma unroll
    for (int ct = 0; ct < 4; ++ct) {
        const int col = colbase0 + ct * 16 + m;
        const float knc = kn[col];
        #pragma unroll
        for (int r = 0; r < 8; ++r) {
            const int row = rowbase + r + 8 * half;
            const float d = 1.0f - acc[ct][r] / fmaxf(xnv[r] * knc, EPS_F);
            dist[(size_t)row * P_N + col] = d;
        }
    }
}

// ---------------------------------------------------------------------------
// Top-5 smallest per row, jax tie-breaking (lowest index on equal value).
// Block-per-row; register insertion lists + deterministic LDS tournament.
// ---------------------------------------------------------------------------
__global__ __launch_bounds__(256) void PromptPool_topk_kernel(
        const float* __restrict__ dist,
        int* __restrict__ idxOut,
        float* __restrict__ rowLoss) {
    const int row = blockIdx.x;
    const int t = threadIdx.x;
    const float* drow = dist + (size_t)row * P_N;

    float lv[K_N];
    int   li[K_N];
    #pragma unroll
    for (int j = 0; j < K_N; ++j) { lv[j] = 3.4e38f; li[j] = 0x7FFFFFFF; }

    for (int c = t; c < P_N; c += 256) {
        const float v = drow[c];
        if (better(v, c, lv[K_N - 1], li[K_N - 1])) {
            lv[K_N - 1] = v; li[K_N - 1] = c;
            #pragma unroll
            for (int j = K_N - 1; j > 0; --j) {
                if (better(lv[j], li[j], lv[j - 1], li[j - 1])) {
                    float tv = lv[j]; lv[j] = lv[j - 1]; lv[j - 1] = tv;
                    int   ti = li[j]; li[j] = li[j - 1]; li[j - 1] = ti;
                }
            }
        }
    }

    __shared__ float sv[256];
    __shared__ int   sc[256];
    __shared__ int   so[256];
    int head = 0;
    float lsum = 0.f;

    for (int k = 0; k < K_N; ++k) {
        sv[t] = (head < K_N) ? lv[head] : 3.4e38f;
        sc[t] = (head < K_N) ? li[head] : 0x7FFFFFFF;
        so[t] = t;
        __syncthreads();
        for (int s = 128; s > 0; s >>= 1) {
            if (t < s) {
                if (better(sv[t + s], sc[t + s], sv[t], sc[t])) {
                    sv[t] = sv[t + s]; sc[t] = sc[t + s]; so[t] = so[t + s];
                }
            }
            __syncthreads();
        }
        if (t == 0) {
            idxOut[row * K_N + k] = sc[0];
            lsum += sv[0];
        }
        const int win = so[0];
        __syncthreads();
        if (t == win) ++head;
        __syncthreads();
    }
    if (t == 0) rowLoss[row] = lsum;
}

// ---------------------------------------------------------------------------
// Gather: prompts[b, k*L+l, d] = values[idx[b,k], l, d].  float4 copies.
// One block per (b, k): 5*1024 floats = 1280 float4.
// ---------------------------------------------------------------------------
__global__ __launch_bounds__(256) void PromptPool_gather_kernel(
        const float* __restrict__ values,
        const int* __restrict__ idxA,
        float* __restrict__ out) {
    const int blk = blockIdx.x;
    const int b = blk / K_N;
    const int k = blk % K_N;
    const int p = idxA[b * K_N + k];
    const float4* src = (const float4*)(values + (size_t)p * L_N * D_N);
    float4* dst = (float4*)(out + (size_t)b * (K_N * L_N * D_N) + (size_t)k * (L_N * D_N));
    #pragma unroll
    for (int i = threadIdx.x; i < (L_N * D_N) / 4; i += 256) dst[i] = src[i];
}

// ---------------------------------------------------------------------------
// Deterministic loss reduction (fixed order, no float atomics).
// ---------------------------------------------------------------------------
__global__ __launch_bounds__(256) void PromptPool_finalize_kernel(
        const float* __restrict__ rowLoss,
        float* __restrict__ out, long loss_off) {
    __shared__ float sv[256];
    const int t = threadIdx.x;
    float sum = 0.f;
    for (int i = t; i < B_N; i += 256) sum += rowLoss[i];
    sv[t] = sum;
    __syncthreads();
    for (int s = 128; s > 0; s >>= 1) {
        if (t < s) sv[t] += sv[t + s];
        __syncthreads();
    }
    if (t == 0) out[loss_off] = sv[0] / (float)(B_N * K_N);
}

// ---------------------------------------------------------------------------
extern "C" void kernel_launch(void* const* d_in, const int* in_sizes, int n_in,
                              void* d_out, int out_size, void* d_ws, size_t ws_size,
                              hipStream_t stream) {
    const float* x      = (const float*)d_in[0];   // (B,1,D)
    const float* keys   = (const float*)d_in[1];   // (P,D)
    const float* values = (const float*)d_in[2];   // (P,L,D)
    float* out = (float*)d_out;                    // prompts flat + loss

    // Workspace layout (all offsets 256B aligned).
    char* ws = (char*)d_ws;
    size_t off = 0;
    float* wsDist = (float*)(ws + off);            off += (size_t)B_N * P_N * sizeof(float);
    unsigned short* xhi = (unsigned short*)(ws + off); off += (size_t)B_N * D_N * 2;
    unsigned short* xlo = (unsigned short*)(ws + off); off += (size_t)B_N * D_N * 2;
    unsigned short* khi = (unsigned short*)(ws + off); off += (size_t)P_N * D_N * 2;
    unsigned short* klo = (unsigned short*)(ws + off); off += (size_t)P_N * D_N * 2;
    float* xn      = (float*)(ws + off);           off += (size_t)B_N * sizeof(float);
    float* kn      = (float*)(ws + off);           off += (size_t)P_N * sizeof(float);
    int*   idxA    = (int*)(ws + off);             off += (size_t)B_N * K_N * sizeof(int);
    float* rowLoss = (float*)(ws + off);           off += (size_t)B_N * sizeof(float);
    (void)ws_size; (void)in_sizes; (void)n_in;

    // 1. Precision-split inputs into bf16 hi/lo pairs.
    PromptPool_split_kernel<<<(B_N * D_N / 4 + 255) / 256, 256, 0, stream>>>(
        x, xhi, xlo, (long)B_N * D_N / 4);
    PromptPool_split_kernel<<<(P_N * D_N / 4 + 255) / 256, 256, 0, stream>>>(
        keys, khi, klo, (long)P_N * D_N / 4);

    // 2. Row norms (wave32 per row).
    PromptPool_norms_kernel<<<(B_N + P_N) / 8, 256, 0, stream>>>(x, keys, xn, kn);

    // 3. WMMA GEMM + cosine distance.
    dim3 ggrid(B_N / 16, P_N / 256);
    PromptPool_gemm_dist_kernel<<<ggrid, 128, 0, stream>>>(
        xhi, xlo, khi, klo, xn, kn, wsDist);

    // 4. Per-row top-5.
    PromptPool_topk_kernel<<<B_N, 256, 0, stream>>>(wsDist, idxA, rowLoss);

    // 5. Gather prompts (HBM-write bound, float4).
    PromptPool_gather_kernel<<<B_N * K_N, 256, 0, stream>>>(values, idxA, out);

    // 6. Deterministic loss mean into the tail scalar.
    PromptPool_finalize_kernel<<<1, 256, 0, stream>>>(
        rowLoss, out, (long)out_size - 1);
}
